// DependencyParseModel_43757126812263
// MI455X (gfx1250) — matrix operand order repaired and checked by reading, not compiled, not run on hardware
//
#include <hip/hip_runtime.h>
#include <hip/hip_bf16.h>
#include <math.h>

// ---------------------------------------------------------------------------
// Dependency-parser forward pass for MI455X (gfx1250, wave32, WMMA).
//   SEQ=256, H=400, 4H=1600, BI=800, MLP_IN=1600, out = 257x257 fp32.
// GEMMs run on v_wmma_f32_16x16x32_bf16 (f32 accumulate) with 16x64 tiles per
// wave (A-fragment reused across 4 accumulators); the sequential LSTM
// recurrence runs as a persistent workgroup per direction with Whh streamed
// from L2; the pairwise tanh-MLP is VALU with LDS-staged A-row/w2 and
// global_prefetch of the next B row.
// ---------------------------------------------------------------------------

typedef __attribute__((ext_vector_type(16))) __bf16 v16bf;
typedef __attribute__((ext_vector_type(8)))  __bf16 v8bf;
typedef __attribute__((ext_vector_type(8)))  float  v8f;

#define SEQ      256
#define HH       400
#define G4       1600
#define BI       800
#define DMLP     1600
#define OUTD     257
#define K0PAD    416   // 400 rounded up to multiple of 32 for WMMA K-loop

__device__ __forceinline__ float sigmoidf_(float x) {
    return 1.0f / (1.0f + __expf(-x));
}

// ---------------------------------------------------------------------------
// 1) Embedding gather + concat -> bf16, padded to K0PAD columns with zeros.
// ---------------------------------------------------------------------------
__global__ void embed_kernel(const int* __restrict__ words,
                             const int* __restrict__ tags,
                             const float* __restrict__ wemb,
                             const float* __restrict__ temb,
                             __bf16* __restrict__ xb) {
    int t = blockIdx.x;
    int w = words[t], tg = tags[t];
    for (int k = threadIdx.x; k < K0PAD; k += blockDim.x) {
        float v = 0.0f;
        if (k < 300)       v = wemb[w * 300 + k];
        else if (k < 400)  v = temb[tg * 100 + (k - 300)];
        xb[t * K0PAD + k] = (__bf16)v;
    }
}

// ---------------------------------------------------------------------------
// 2) fp32 -> bf16 row convert with column offset + zero-padding.
// ---------------------------------------------------------------------------
__global__ void convert_bf16_kernel(const float* __restrict__ src,
                                    __bf16* __restrict__ dst,
                                    int Ksrc, int srcld, int koff, int Kp) {
    int row = blockIdx.x;
    for (int k = threadIdx.x; k < Kp; k += blockDim.x) {
        float v = (k < Ksrc) ? src[row * srcld + koff + k] : 0.0f;
        dst[row * Kp + k] = (__bf16)v;
    }
}

// ---------------------------------------------------------------------------
// 3) WMMA GEMM:  C[M,N] = X[M,K](bf16) * W[N,K](bf16)^T (+ bias[N]) in fp32.
//    One 16x64 output tile per wave (4 f32 accumulators share one A-frag);
//    K stepped by 32.
//    A-frag (16x32 bf16): lane<16 holds row M=lane, K
//      {k0..k0+15}; lane>=16 holds row lane-16,
//      {k0+16..k0+31}  -> two 16B loads per lane.
//    B-frag (32x16, SWMMAC-style column-major dense B): lane&15 = column N,
//      lane>>4 selects K half -> one contiguous 32B load per lane.
// ---------------------------------------------------------------------------
__global__ void gemm_bf16_wmma(const __bf16* __restrict__ X, int ldx,
                               const __bf16* __restrict__ W, int ldw,
                               const float* __restrict__ bias,
                               float* __restrict__ C, int ldc,
                               int M, int N, int K) {
    const int lane = threadIdx.x & 31;
    const int wave = threadIdx.x >> 5;
    const int wavesPerBlock = blockDim.x >> 5;
    const int tile   = blockIdx.x * wavesPerBlock + wave;
    const int mtiles = M >> 4;
    const int ntile  = tile / mtiles;       // 64-wide N tile index
    const int mtile  = tile % mtiles;
    if (ntile * 64 >= N) return;

    const int half  = lane >> 4;            // 0: low K half, 1: high K half
    const int r     = (mtile << 4) + (lane & 15);
    const int nbase = (ntile << 6) + (lane & 15);

    const __bf16* xrow = X + (size_t)r * ldx + half * 16;
    const __bf16* wr0 = W + (size_t)(nbase     ) * ldw + half * 16;
    const __bf16* wr1 = W + (size_t)(nbase + 16) * ldw + half * 16;
    const __bf16* wr2 = W + (size_t)(nbase + 32) * ldw + half * 16;
    const __bf16* wr3 = W + (size_t)(nbase + 48) * ldw + half * 16;

    v8f acc0 = {}, acc1 = {}, acc2 = {}, acc3 = {};
    for (int k0 = 0; k0 < K; k0 += 32) {
        v8bf a0 = *(const v8bf*)(xrow + k0);
        v8bf a1 = *(const v8bf*)(xrow + k0 + 8);
        v16bf a = __builtin_shufflevector(a0, a1,
                    0, 1, 2, 3, 4, 5, 6, 7, 8, 9, 10, 11, 12, 13, 14, 15);
        v16bf b0 = *(const v16bf*)(wr0 + k0);
        v16bf b1 = *(const v16bf*)(wr1 + k0);
        v16bf b2 = *(const v16bf*)(wr2 + k0);
        v16bf b3 = *(const v16bf*)(wr3 + k0);
        acc0 = __builtin_amdgcn_wmma_f32_16x16x32_bf16(
                   false, a, false, b0, (short)0, acc0, false, false);
        acc1 = __builtin_amdgcn_wmma_f32_16x16x32_bf16(
                   false, a, false, b1, (short)0, acc1, false, false);
        acc2 = __builtin_amdgcn_wmma_f32_16x16x32_bf16(
                   false, a, false, b2, (short)0, acc2, false, false);
        acc3 = __builtin_amdgcn_wmma_f32_16x16x32_bf16(
                   false, a, false, b3, (short)0, acc3, false, false);
    }

    const int mrow = (mtile << 4) + (half << 3);
    v8f accs[4] = {acc0, acc1, acc2, acc3};
#pragma unroll
    for (int sub = 0; sub < 4; ++sub) {
        const int n = nbase + (sub << 4);
        const float bv = bias ? bias[n] : 0.0f;
#pragma unroll
        for (int v = 0; v < 8; ++v)
            C[(size_t)(mrow + v) * ldc + n] = accs[sub][v] + bv;
    }
}

// ---------------------------------------------------------------------------
// 4) Persistent bidirectional LSTM layer.  grid = 2 blocks (dir 0 fwd, 1 bwd).
//    Gpre[t,g] = x[t] @ Wih^T + b (precomputed by WMMA GEMM).
//    Per step: gates[g] = Gpre[t,g] + Whh[g,:] . h_prev  (float4 GEMV from L2),
//    then torch gate order i,f,g,o; writes h into out[t*outStride + dir*H + j].
// ---------------------------------------------------------------------------
__global__ void lstm_layer_kernel(const float* __restrict__ Gf,
                                  const float* __restrict__ Gb,
                                  const float* __restrict__ Whhf,
                                  const float* __restrict__ Whhb,
                                  const float* __restrict__ h0base,
                                  const float* __restrict__ c0base,
                                  float* __restrict__ out, int outStride) {
    __shared__ __align__(16) float hprev[HH];
    __shared__ __align__(16) float cst[HH];
    __shared__ __align__(16) float gates[G4];

    const int dir = blockIdx.x;
    const float* Gpre = dir ? Gb : Gf;
    const float* Whh  = dir ? Whhb : Whhf;
    const float* h0 = h0base + dir * HH;
    const float* c0 = c0base + dir * HH;
    float* o = out + dir * HH;

    for (int j = threadIdx.x; j < HH; j += blockDim.x) {
        hprev[j] = h0[j];
        cst[j]   = c0[j];
    }
    __syncthreads();

    for (int s = 0; s < SEQ; ++s) {
        const int t = dir ? (SEQ - 1 - s) : s;
        // phase 1: gate pre-activations
        for (int g = threadIdx.x; g < G4; g += blockDim.x) {
            const float4* wr = (const float4*)(Whh + (size_t)g * HH);
            const float4* hp = (const float4*)hprev;
            float a0 = 0.f, a1 = 0.f, a2 = 0.f, a3 = 0.f;
#pragma unroll 4
            for (int k = 0; k < HH / 4; ++k) {
                float4 wv = wr[k], hv = hp[k];
                a0 += wv.x * hv.x; a1 += wv.y * hv.y;
                a2 += wv.z * hv.z; a3 += wv.w * hv.w;
            }
            gates[g] = Gpre[(size_t)t * G4 + g] + ((a0 + a1) + (a2 + a3));
        }
        __syncthreads();
        // phase 2: nonlinearities + state update (torch order i,f,g,o)
        for (int j = threadIdx.x; j < HH; j += blockDim.x) {
            float ig = sigmoidf_(gates[j]);
            float fg = sigmoidf_(gates[HH + j]);
            float gg = tanhf(gates[2 * HH + j]);
            float og = sigmoidf_(gates[3 * HH + j]);
            float c  = fg * cst[j] + ig * gg;
            float h  = og * tanhf(c);
            cst[j]   = c;
            hprev[j] = h;
            o[(size_t)t * outStride + j] = h;
        }
        __syncthreads();
    }
}

// ---------------------------------------------------------------------------
// 5) Zero the whole 257x257 output (borders stay zero; inner filled next).
// ---------------------------------------------------------------------------
__global__ void zero_kernel(float* __restrict__ p, int n) {
    int i = blockIdx.x * blockDim.x + threadIdx.x;
    if (i < n) p[i] = 0.0f;
}

// ---------------------------------------------------------------------------
// 6) Pairwise arc scores. A already includes b1m.
//    out[i+1, j+1] = (i==j) ? 0 : w2 . tanh(A[i] + B[j]) + b2.
//    One block per i; A[i] and w2 staged in LDS; next B row prefetched.
// ---------------------------------------------------------------------------
__global__ void pairwise_kernel(const float* __restrict__ A,
                                const float* __restrict__ Bm,
                                const float* __restrict__ w2,
                                const float* __restrict__ b2p,
                                float* __restrict__ out) {
    __shared__ float Ai[DMLP];
    __shared__ float w2s[DMLP];
    __shared__ float wsum[8];

    const int i = blockIdx.x;
    for (int k = threadIdx.x; k < DMLP; k += blockDim.x) {
        Ai[k]  = A[(size_t)i * DMLP + k];
        w2s[k] = w2[k];
    }
    __syncthreads();

    const float b2 = b2p[0];
    const int lane = threadIdx.x & 31;
    const int wave = threadIdx.x >> 5;
    const int nwaves = blockDim.x >> 5;

    for (int j = 0; j < SEQ; ++j) {
        const float* Bj = Bm + (size_t)j * DMLP;
        if (j + 1 < SEQ)
            __builtin_prefetch(Bm + (size_t)(j + 1) * DMLP + threadIdx.x, 0, 0);
        float p = 0.0f;
        for (int k = threadIdx.x; k < DMLP; k += blockDim.x)
            p += w2s[k] * tanhf(Ai[k] + Bj[k]);
        for (int off = 16; off > 0; off >>= 1)
            p += __shfl_down(p, off, 32);
        if (lane == 0) wsum[wave] = p;
        __syncthreads();
        if (threadIdx.x == 0) {
            float sres = b2;
            for (int ww = 0; ww < nwaves; ++ww) sres += wsum[ww];
            if (i == j) sres = 0.0f;
            out[(size_t)(i + 1) * OUTD + (j + 1)] = sres;
        }
        __syncthreads();
    }
}

// ---------------------------------------------------------------------------
// Host orchestration
// ---------------------------------------------------------------------------
extern "C" void kernel_launch(void* const* d_in, const int* in_sizes, int n_in,
                              void* d_out, int out_size, void* d_ws, size_t ws_size,
                              hipStream_t stream) {
    (void)in_sizes; (void)n_in; (void)out_size; (void)ws_size;

    const int*   words = (const int*)d_in[0];
    const int*   tags  = (const int*)d_in[1];
    /* d_in[2] = heads (int64) — unused by the reference output */
    const float* wemb  = (const float*)d_in[3];
    const float* temb  = (const float*)d_in[4];
    const float* Wih0f = (const float*)d_in[5];
    const float* Whh0f = (const float*)d_in[6];
    const float* b0f   = (const float*)d_in[7];
    const float* Wih0b = (const float*)d_in[8];
    const float* Whh0b = (const float*)d_in[9];
    const float* b0b   = (const float*)d_in[10];
    const float* Wih1f = (const float*)d_in[11];
    const float* Whh1f = (const float*)d_in[12];
    const float* b1f   = (const float*)d_in[13];
    const float* Wih1b = (const float*)d_in[14];
    const float* Whh1b = (const float*)d_in[15];
    const float* b1b   = (const float*)d_in[16];
    const float* h0    = (const float*)d_in[17];
    const float* c0    = (const float*)d_in[18];
    const float* W1    = (const float*)d_in[19];
    const float* b1m   = (const float*)d_in[20];
    const float* w2    = (const float*)d_in[21];
    const float* b2m   = (const float*)d_in[22];
    float* out = (float*)d_out;

    // workspace allocator (256B aligned)
    char* ws = (char*)d_ws;
    size_t off = 0;
    auto alloc = [&](size_t bytes) -> void* {
        void* p = ws + off;
        off = (off + bytes + 255) & ~(size_t)255;
        return p;
    };

    __bf16* xb   = (__bf16*)alloc((size_t)SEQ * K0PAD * 2);
    __bf16* W0fb = (__bf16*)alloc((size_t)G4 * K0PAD * 2);
    __bf16* W0bb = (__bf16*)alloc((size_t)G4 * K0PAD * 2);
    __bf16* W1fb = (__bf16*)alloc((size_t)G4 * BI * 2);
    __bf16* W1bb = (__bf16*)alloc((size_t)G4 * BI * 2);
    __bf16* WLb  = (__bf16*)alloc((size_t)DMLP * BI * 2);
    __bf16* WRb  = (__bf16*)alloc((size_t)DMLP * BI * 2);
    float*  G0f  = (float*)alloc((size_t)SEQ * G4 * 4);
    float*  G0b  = (float*)alloc((size_t)SEQ * G4 * 4);
    float*  G1f  = (float*)alloc((size_t)SEQ * G4 * 4);
    float*  G1b  = (float*)alloc((size_t)SEQ * G4 * 4);
    float*  x1   = (float*)alloc((size_t)SEQ * BI * 4);
    __bf16* x1b  = (__bf16*)alloc((size_t)SEQ * BI * 2);
    float*  h1   = (float*)alloc((size_t)SEQ * BI * 4);
    __bf16* h1b  = (__bf16*)alloc((size_t)SEQ * BI * 2);
    float*  Aa   = (float*)alloc((size_t)SEQ * DMLP * 4);
    float*  Bb   = (float*)alloc((size_t)SEQ * DMLP * 4);

    // 1) embeddings -> bf16 (padded K)
    embed_kernel<<<SEQ, 256, 0, stream>>>(words, tags, wemb, temb, xb);

    // 2) weight conversions to bf16 (layer0 K padded to 416)
    convert_bf16_kernel<<<G4, 256, 0, stream>>>(Wih0f, W0fb, 400, 400, 0, K0PAD);
    convert_bf16_kernel<<<G4, 256, 0, stream>>>(Wih0b, W0bb, 400, 400, 0, K0PAD);
    convert_bf16_kernel<<<G4, 256, 0, stream>>>(Wih1f, W1fb, BI, BI, 0, BI);
    convert_bf16_kernel<<<G4, 256, 0, stream>>>(Wih1b, W1bb, BI, BI, 0, BI);
    convert_bf16_kernel<<<DMLP, 256, 0, stream>>>(W1, WLb, BI, DMLP, 0,  BI);
    convert_bf16_kernel<<<DMLP, 256, 0, stream>>>(W1, WRb, BI, DMLP, BI, BI);

    // GEMM grid: (M/16)*(N/64) tiles, 8 waves (256 threads) per block
    const int tiles   = (SEQ / 16) * (G4 / 64);   // 16 * 25 = 400
    const int gblocks = tiles / 8;                // 50

    // 3) layer-0 input projections (WMMA)
    gemm_bf16_wmma<<<gblocks, 256, 0, stream>>>(xb, K0PAD, W0fb, K0PAD, b0f,
                                                G0f, G4, SEQ, G4, K0PAD);
    gemm_bf16_wmma<<<gblocks, 256, 0, stream>>>(xb, K0PAD, W0bb, K0PAD, b0b,
                                                G0b, G4, SEQ, G4, K0PAD);

    // 4) layer-0 recurrence (persistent, one workgroup per direction)
    lstm_layer_kernel<<<2, 512, 0, stream>>>(G0f, G0b, Whh0f, Whh0b,
                                             h0 + 0 * HH, c0 + 0 * HH, x1, BI);

    // 5) x1 -> bf16, layer-1 input projections
    convert_bf16_kernel<<<SEQ, 256, 0, stream>>>(x1, x1b, BI, BI, 0, BI);
    gemm_bf16_wmma<<<gblocks, 256, 0, stream>>>(x1b, BI, W1fb, BI, b1f,
                                                G1f, G4, SEQ, G4, BI);
    gemm_bf16_wmma<<<gblocks, 256, 0, stream>>>(x1b, BI, W1bb, BI, b1b,
                                                G1b, G4, SEQ, G4, BI);

    // 6) layer-1 recurrence
    lstm_layer_kernel<<<2, 512, 0, stream>>>(G1f, G1b, Whh1f, Whh1b,
                                             h0 + 2 * HH, c0 + 2 * HH, h1, BI);

    // 7) arc-MLP projections: A = h@W1[:, :800]^T + b1m ; B = h@W1[:, 800:]^T
    convert_bf16_kernel<<<SEQ, 256, 0, stream>>>(h1, h1b, BI, BI, 0, BI);
    gemm_bf16_wmma<<<gblocks, 256, 0, stream>>>(h1b, BI, WLb, BI, b1m,
                                                Aa, DMLP, SEQ, DMLP, BI);
    gemm_bf16_wmma<<<gblocks, 256, 0, stream>>>(h1b, BI, WRb, BI, nullptr,
                                                Bb, DMLP, SEQ, DMLP, BI);

    // 8) output: zero borders, fill pairwise scores
    zero_kernel<<<(OUTD * OUTD + 255) / 256, 256, 0, stream>>>(out, OUTD * OUTD);
    pairwise_kernel<<<SEQ, 256, 0, stream>>>(Aa, Bb, w2, b2m, out);
}
